// TokenCrossAttention_25271587569753
// MI455X (gfx1250) — matrix operand order repaired and checked by reading, not compiled, not run on hardware
//
#include <hip/hip_runtime.h>

// Problem constants (fixed by the reference)
#define BATCH 4
#define SEQ   2048
#define CH    1024
#define HEADS 16
#define HD    64
#define EPSV  1e-6f

typedef __bf16 bf16;
typedef __attribute__((ext_vector_type(16))) __bf16 v16bf;
typedef __attribute__((ext_vector_type(8)))  __bf16 v8bf;
typedef __attribute__((ext_vector_type(8)))  float  v8f;
typedef __attribute__((ext_vector_type(4)))  int    v4i;

// ---- gfx1250 async-copy (guarded: never breaks compilation) ----------------
#if defined(__has_builtin)
#  if __has_builtin(__builtin_amdgcn_global_load_async_to_lds_b128) && \
      __has_builtin(__builtin_amdgcn_s_wait_asynccnt)
#    define USE_ASYNC_COPY 1
#  endif
#endif
#ifndef USE_ASYNC_COPY
#  define USE_ASYNC_COPY 0
#endif

typedef __attribute__((address_space(1))) v4i gv4i;   // __global int4
typedef __attribute__((address_space(3))) v4i lv4i;   // __shared__ int4
static __device__ __forceinline__ gv4i* to_gbl(const void* p) {
    return (gv4i*)(unsigned long long)p;
}
static __device__ __forceinline__ lv4i* to_lds(void* p) {
    // low 32 bits of a generic LDS pointer are the AS(3) byte address
    return (lv4i*)(unsigned)(unsigned long long)p;
}

// Raw HW exp2 (v_exp_f32): no denormal guard, no extra VALU. Sub-denormal
// probabilities flush to zero, ~120 orders below the EPS smoothing floor.
static __device__ __forceinline__ float fast_exp2(float x) {
#if defined(__has_builtin) && __has_builtin(__builtin_amdgcn_exp2f)
    return __builtin_amdgcn_exp2f(x);
#else
    float r;
    __asm__("v_exp_f32 %0, %1" : "=v"(r) : "v"(x));
    return r;
#endif
}

// Build a 16-element bf16 WMMA fragment from two contiguous 16-byte chunks.
static __device__ __forceinline__ v16bf make_frag(const bf16* p0, const bf16* p1) {
    v8bf a = *(const v8bf*)p0;
    v8bf b = *(const v8bf*)p1;
    v16bf r;
#pragma unroll
    for (int i = 0; i < 8; ++i) { r[i] = a[i]; r[8 + i] = b[i]; }
    return r;
}

static __device__ __forceinline__ v8f wmma_bf16(v16bf a, v16bf b, v8f c) {
    return __builtin_amdgcn_wmma_f32_16x16x32_bf16(false, a, false, b, (short)0, c,
                                                   false, false);
}

// ---------------------------------------------------------------- convert ---
__global__ void cvt_f32_to_bf16(const float* __restrict__ in, bf16* __restrict__ out,
                                size_t n) {
    size_t i = (size_t)blockIdx.x * blockDim.x + threadIdx.x;
    size_t stride = (size_t)gridDim.x * blockDim.x;
    for (; i < n; i += stride) out[i] = (bf16)in[i];
}

// ------------------------------------------------------------------- GEMM ---
// C[M x Nn] = cscale * A[M x K] * B[K x Nn]  (row-major bf16, f32 accumulate)
// Block: 256 threads = 8 waves (4 row-groups x 2 col-groups).
// Block tile 128x128; wave tile 32x64 -> 8 wmma per 32-wide K step.
// Double-buffered LDS: stage tile k+1 (async A DMA + packed B transpose)
// while computing tile k; one barrier per K step.
// MODE 0: bf16 row-major out. MODE 1: f32 row-major out.
// MODE 2: bf16 scattered to VT layout (B,H,HD,SEQ) (for the V matrix).
template <int MODE>
__global__ __launch_bounds__(256)
void gemm_bf16(const bf16* __restrict__ A, const bf16* __restrict__ Bm,
               void* __restrict__ outp, int M, int Nn, int K, float cscale) {
    __shared__ bf16 At[2][128][32];          // A tiles, rows of 64B
    __shared__ bf16 BT[2][128][40];          // B tiles transposed, padded rows (80B)

    const int tid  = threadIdx.x;
    const int wave = tid >> 5;
    const int lane = tid & 31;
    const int l16  = lane & 15;
    const int hi   = lane >> 4;
    const int wr   = wave >> 1;              // row group: 32 rows
    const int wc   = wave & 1;               // col group: 64 cols

    const int rowBase = blockIdx.x * 128;
    const int colBase = blockIdx.y * 128;

    auto stage = [&](int k0, int sb) {
        // A tile 128x32 (async DMA to LDS when available)
#pragma unroll
        for (int i = 0; i < 2; ++i) {
            int idx = i * 256 + tid;          // 0..511 chunks of 16B
            int row = idx >> 2;
            int seg = (idx & 3) * 8;
            const bf16* g = A + (size_t)(rowBase + row) * K + k0 + seg;
#if USE_ASYNC_COPY
            __builtin_amdgcn_global_load_async_to_lds_b128(
                to_gbl(g), to_lds(&At[sb][row][seg]), 0, 0);
#else
            *(v8bf*)&At[sb][row][seg] = *(const v8bf*)g;
#endif
        }
        // B tile 32x128 transposed; packed dword LDS stores
#pragma unroll
        for (int i = 0; i < 4; ++i) {
            int idx = i * 256 + tid;          // 0..1023 (row-pair x col-pair)
            int kp  = (idx >> 6) * 2;         // 0,2,..,30
            int n   = (idx & 63) * 2;         // 0..126
            const bf16* gp = &Bm[(size_t)(k0 + kp) * Nn + colBase + n];
            unsigned va = *(const unsigned*)gp;        // row kp  : cols n, n+1
            unsigned vb = *(const unsigned*)(gp + Nn); // row kp+1: cols n, n+1
            *(unsigned*)&BT[sb][n][kp]     = (va & 0xFFFFu) | (vb << 16);
            *(unsigned*)&BT[sb][n + 1][kp] = (va >> 16) | (vb & 0xFFFF0000u);
        }
    };

    v8f acc[2][4];
#pragma unroll
    for (int rr = 0; rr < 2; ++rr)
#pragma unroll
        for (int t = 0; t < 4; ++t) { v8f z = {}; acc[rr][t] = z; }

    stage(0, 0);
#if USE_ASYNC_COPY
    __builtin_amdgcn_s_wait_asynccnt(0);
#endif
    __syncthreads();

    int buf = 0;
    for (int k0 = 0; k0 < K; k0 += 32) {
        if (k0 + 32 < K) stage(k0 + 32, buf ^ 1);   // prefetch next tile

        v16bf afrag[2];
#pragma unroll
        for (int rr = 0; rr < 2; ++rr) {
            const bf16* ar = &At[buf][wr * 32 + rr * 16 + l16][hi ? 8 : 0];
            afrag[rr] = make_frag(ar, ar + 16);
        }
#pragma unroll
        for (int t = 0; t < 4; ++t) {
            const bf16* br = &BT[buf][wc * 64 + t * 16 + l16][hi ? 8 : 0];
            v16bf bfrag = make_frag(br, br + 16);
            acc[0][t] = wmma_bf16(afrag[0], bfrag, acc[0][t]);
            acc[1][t] = wmma_bf16(afrag[1], bfrag, acc[1][t]);
        }
#if USE_ASYNC_COPY
        __builtin_amdgcn_s_wait_asynccnt(0);
#endif
        __syncthreads();
        buf ^= 1;
    }

#pragma unroll
    for (int rr = 0; rr < 2; ++rr) {
#pragma unroll
        for (int t = 0; t < 4; ++t) {
            const int col = colBase + wc * 64 + t * 16 + l16;
#pragma unroll
            for (int r = 0; r < 8; ++r) {
                const int row = rowBase + wr * 32 + rr * 16 + r + hi * 8;
                float val = acc[rr][t][r] * cscale;
                if (MODE == 0) {
                    ((bf16*)outp)[(size_t)row * Nn + col] = (bf16)val;
                } else if (MODE == 1) {
                    ((float*)outp)[(size_t)row * Nn + col] = val;
                } else {
                    int b = row >> 11, n = row & (SEQ - 1);
                    int h = col >> 6,  dd = col & (HD - 1);
                    ((bf16*)outp)[(((size_t)(b * HEADS + h)) * HD + dd) * SEQ + n] =
                        (bf16)val;
                }
            }
        }
    }
}

// ------------------------------------------------------------------- sumV ---
__global__ void sumv_kernel(const bf16* __restrict__ VT, float* __restrict__ sumv) {
    int bh = blockIdx.x;            // B*H
    int dd = threadIdx.x;           // HD threads
    const bf16* r = VT + ((size_t)bh * HD + dd) * SEQ;
    float s = 0.f;
    for (int n = 0; n < SEQ; ++n) s += (float)r[n];
    sumv[bh * HD + dd] = s;
}

// --------------------------------------------------------- flash attention ---
// Grid: (SEQ/256, B*H). 256 threads = 8 waves; wave w owns 32 query rows
// (2 q-tiles), so K/V fragments are amortized over 2x the WMMA work.
// Q is pre-scaled by d^-0.5 * log2(e), so P = exp2(S)*fuse maps directly onto
// v_exp_f32 (HW exp is 2^x) with no per-element multiply. Zero-shift softmax:
// logits are |a| <~ 15 here, exp2 cannot overflow, and the reference's
// (e+EPS/N)/(Se+EPS) smoothing differs only in the ~1e-6 EPS terms under a
// shift change. Row-sum of P comes from the ones-column WMMA (O4 = P*1).
__global__ __launch_bounds__(256)
void flash_attn(const bf16* __restrict__ Q, const bf16* __restrict__ Km,
                const bf16* __restrict__ VT, const float* __restrict__ decisions,
                const float* __restrict__ sumv, bf16* __restrict__ attnOut) {
    __shared__ bf16 pls[8][2][16][40];       // per-wave, per-q-tile P patch

    const int tid  = threadIdx.x;
    const int wave = tid >> 5;
    const int lane = tid & 31;
    const int l16  = lane & 15;
    const int hi   = lane >> 4;

    const int bh = blockIdx.y;
    const int b  = bh >> 4;
    const int h  = bh & 15;
    const int qBase = blockIdx.x * 256 + wave * 32;

    v16bf qf[2][2];
#pragma unroll
    for (int qt = 0; qt < 2; ++qt) {
        const bf16* qrow = Q + ((size_t)(b * SEQ + qBase + qt * 16 + l16)) * CH +
                           h * HD + (hi ? 8 : 0);
        qf[qt][0] = make_frag(qrow,      qrow + 16);
        qf[qt][1] = make_frag(qrow + 32, qrow + 48);
    }

    const bf16* kbase  = Km + ((size_t)b * SEQ) * CH + h * HD;
    const bf16* vtbase = VT + ((size_t)bh * HD) * SEQ;
    const float* dec   = decisions + (size_t)b * SEQ;

    v16bf ones;
#pragma unroll
    for (int i = 0; i < 16; ++i) ones[i] = (bf16)1.0f;

    v8f O[2][4], O4[2];
#pragma unroll
    for (int qt = 0; qt < 2; ++qt) {
        { v8f z = {}; O4[qt] = z; }
#pragma unroll
        for (int t = 0; t < 4; ++t) { v8f z = {}; O[qt][t] = z; }
    }

    for (int kb = 0; kb < SEQ; kb += 32) {
        // ---- K^T fragments (shared by both q-tiles) ----
        const bf16* kr0 = kbase + (size_t)(kb + l16) * CH + (hi ? 8 : 0);
        v16bf kf00 = make_frag(kr0,      kr0 + 16);
        v16bf kf01 = make_frag(kr0 + 32, kr0 + 48);
        const bf16* kr1 = kbase + (size_t)(kb + 16 + l16) * CH + (hi ? 8 : 0);
        v16bf kf10 = make_frag(kr1,      kr1 + 16);
        v16bf kf11 = make_frag(kr1 + 32, kr1 + 48);

        const float fuse0 = 1.0f - dec[kb + l16];
        const float fuse1 = 1.0f - dec[kb + 16 + l16];

        // ---- S = Q K^T, then P = exp2(S) * fuse, written transposed ----
#pragma unroll
        for (int qt = 0; qt < 2; ++qt) {
            v8f z = {};
            v8f s0 = wmma_bf16(qf[qt][0], kf00, z);
            s0     = wmma_bf16(qf[qt][1], kf01, s0);
            v8f s1 = wmma_bf16(qf[qt][0], kf10, z);
            s1     = wmma_bf16(qf[qt][1], kf11, s1);
#pragma unroll
            for (int r = 0; r < 8; ++r) {
                float e0 = fast_exp2(s0[r]) * fuse0;
                float e1 = fast_exp2(s1[r]) * fuse1;
                pls[wave][qt][r + hi * 8][l16]      = (bf16)e0;
                pls[wave][qt][r + hi * 8][16 + l16] = (bf16)e1;
            }
        }
        __asm__ volatile("" ::: "memory");   // per-wave LDS ops are in-order in HW

        v16bf pf[2];
#pragma unroll
        for (int qt = 0; qt < 2; ++qt) {
            const bf16* pr = &pls[wave][qt][l16][hi ? 8 : 0];
            pf[qt] = make_frag(pr, pr + 16);
        }

        // ---- O += P V (V fragments shared); row-sum via ones WMMA ----
#pragma unroll
        for (int t = 0; t < 4; ++t) {
            const bf16* vr = vtbase + (size_t)(t * 16 + l16) * SEQ + kb + (hi ? 8 : 0);
            v16bf vf = make_frag(vr, vr + 16);
            O[0][t] = wmma_bf16(pf[0], vf, O[0][t]);
            O[1][t] = wmma_bf16(pf[1], vf, O[1][t]);
        }
        O4[0] = wmma_bf16(pf[0], ones, O4[0]);
        O4[1] = wmma_bf16(pf[1], ones, O4[1]);
        __asm__ volatile("" ::: "memory");
    }

    // ---- epilogue: out = (O + (EPS/N) * sumV) / (Sum + EPS), store bf16 ----
    const float epsn = EPSV / (float)SEQ;
#pragma unroll
    for (int qt = 0; qt < 2; ++qt) {
#pragma unroll
        for (int r = 0; r < 8; ++r) {
            float denom = 1.0f / (O4[qt][r] + EPSV);
            int row = qBase + qt * 16 + r + hi * 8;
#pragma unroll
            for (int t = 0; t < 4; ++t) {
                float sv = sumv[bh * HD + t * 16 + l16];
                float o  = (O[qt][t][r] + epsn * sv) * denom;
                attnOut[((size_t)(b * SEQ + row)) * CH + h * HD + t * 16 + l16] = (bf16)o;
            }
        }
    }
}

// ------------------------------------------------------------------ launch ---
extern "C" void kernel_launch(void* const* d_in, const int* in_sizes, int n_in,
                              void* d_out, int out_size, void* d_ws, size_t ws_size,
                              hipStream_t stream) {
    const float* x    = (const float*)d_in[0];
    const float* xo   = (const float*)d_in[1];
    const float* dec  = (const float*)d_in[2];
    const float* Wq   = (const float*)d_in[3];
    const float* Wk   = (const float*)d_in[4];
    const float* Wv   = (const float*)d_in[5];
    const float* Wp   = (const float*)d_in[6];

    const size_t X_ELE = (size_t)BATCH * SEQ * CH;   // 8,388,608
    const size_t W_ELE = (size_t)CH * CH;            // 1,048,576

    char* w = (char*)d_ws;
    bf16* xbf   = (bf16*)w;                w += X_ELE * sizeof(bf16);
    bf16* xobf  = (bf16*)w;                w += X_ELE * sizeof(bf16);
    bf16* Wqbf  = (bf16*)w;                w += W_ELE * sizeof(bf16);
    bf16* Wkbf  = (bf16*)w;                w += W_ELE * sizeof(bf16);
    bf16* Wvbf  = (bf16*)w;                w += W_ELE * sizeof(bf16);
    bf16* Wpbf  = (bf16*)w;                w += W_ELE * sizeof(bf16);
    bf16* Qbf   = (bf16*)w;                w += X_ELE * sizeof(bf16);
    bf16* Kbf   = (bf16*)w;                w += X_ELE * sizeof(bf16);
    bf16* VTbf  = (bf16*)w;                w += X_ELE * sizeof(bf16);
    float* sumv = (float*)w;               w += (size_t)BATCH * HEADS * HD * sizeof(float);
    bf16* attnbf = xbf;   // xbf is dead after the Q projection; reuse it

    dim3 blk(256);

    // 1) fp32 -> bf16 conversions
    cvt_f32_to_bf16<<<dim3(2048), blk, 0, stream>>>(x,  xbf,  X_ELE);
    cvt_f32_to_bf16<<<dim3(2048), blk, 0, stream>>>(xo, xobf, X_ELE);
    cvt_f32_to_bf16<<<dim3(512),  blk, 0, stream>>>(Wq, Wqbf, W_ELE);
    cvt_f32_to_bf16<<<dim3(512),  blk, 0, stream>>>(Wk, Wkbf, W_ELE);
    cvt_f32_to_bf16<<<dim3(512),  blk, 0, stream>>>(Wv, Wvbf, W_ELE);
    cvt_f32_to_bf16<<<dim3(512),  blk, 0, stream>>>(Wp, Wpbf, W_ELE);

    const int M = BATCH * SEQ;               // 8192
    dim3 ggrid(M / 128, CH / 128);           // 64 x 8

    // 2) projections; Q pre-scaled by d^-0.5 * log2(e) so flash uses exp2
    const float qscale = 0.125f * 1.44269504088896340736f;
    gemm_bf16<0><<<ggrid, blk, 0, stream>>>(xbf,  Wqbf, (void*)Qbf,  M, CH, CH, qscale);
    gemm_bf16<0><<<ggrid, blk, 0, stream>>>(xobf, Wkbf, (void*)Kbf,  M, CH, CH, 1.0f);
    gemm_bf16<2><<<ggrid, blk, 0, stream>>>(xobf, Wvbf, (void*)VTbf, M, CH, CH, 1.0f);

    // 3) per-head column sums of V (for the EPS/N smoothing term)
    sumv_kernel<<<dim3(BATCH * HEADS), dim3(HD), 0, stream>>>(VTbf, sumv);

    // 4) flash attention with policy-masked softmax (zero-shift, ones-sum)
    flash_attn<<<dim3(SEQ / 256, BATCH * HEADS), blk, 0, stream>>>(
        Qbf, Kbf, VTbf, dec, sumv, attnbf);

    // 5) output projection -> fp32 d_out
    gemm_bf16<1><<<ggrid, blk, 0, stream>>>(attnbf, Wpbf, d_out, M, CH, CH, 1.0f);
}
